// ScaledDotProductAttention_15908558864399
// MI455X (gfx1250) — matrix operand order repaired and checked by reading, not compiled, not run on hardware
//
#include <hip/hip_runtime.h>

typedef __attribute__((ext_vector_type(16))) __bf16 v16bf;
typedef __attribute__((ext_vector_type(8)))  float  v8f;

#define BATCH 16
#define SEQ   2048
#define DK    64
#define WAVES 4
#define TILE_M 64
#define NTILES (SEQ / TILE_M)
#define NSUB   (TILE_M / 16)
#define TILE_ELEMS (TILE_M * DK)          // 4096 elems, 8KB in bf16

// ---- 16-lane butterfly reductions via DPP (pure VALU, no LDS) ----------------
template <int CTRL>
__device__ __forceinline__ float dpp_mov(float v) {
    int x = __builtin_bit_cast(int, v);
    int y = __builtin_amdgcn_update_dpp(x, x, CTRL, 0xF, 0xF, true);
    return __builtin_bit_cast(float, y);
}
__device__ __forceinline__ float red16_max(float v) {
    v = fmaxf(v, dpp_mov<0x0B1>(v));
    v = fmaxf(v, dpp_mov<0x04E>(v));
    v = fmaxf(v, dpp_mov<0x141>(v));
    v = fmaxf(v, dpp_mov<0x140>(v));
    return v;
}
__device__ __forceinline__ float red16_sum(float v) {
    v += dpp_mov<0x0B1>(v);
    v += dpp_mov<0x04E>(v);
    v += dpp_mov<0x141>(v);
    v += dpp_mov<0x140>(v);
    return v;
}

// =============================================================================
// Pre-pass: K (f32) -> bf16 row-major tiles ; V (f32) -> bf16 V^T tiles (d-major)
// One block per (b, mt) 64x64 tile. blockIdx.x = b*NTILES + mt.
// =============================================================================
__global__ __launch_bounds__(128)
void convert_kv_kernel(const float* __restrict__ K, const float* __restrict__ V,
                       __bf16* __restrict__ wsK, __bf16* __restrict__ wsVT)
{
    __shared__ __align__(64) __bf16 sT[TILE_ELEMS];      // V^T staging
    const int tid = threadIdx.x;
    const size_t tileBase = (size_t)blockIdx.x * TILE_ELEMS;

    const float* kg = K + tileBase;
    const float* vg = V + tileBase;
    __bf16* kd = wsK + tileBase;
#pragma unroll
    for (int i = 0; i < 8; ++i) {
        const int e = tid * 4 + i * 512;
        const float4 kq = *(const float4*)(kg + e);
        kd[e + 0] = (__bf16)kq.x; kd[e + 1] = (__bf16)kq.y;
        kd[e + 2] = (__bf16)kq.z; kd[e + 3] = (__bf16)kq.w;
        const float4 vq = *(const float4*)(vg + e);
        const int r = e >> 6, c = e & 63;                // V[r][c..c+3] -> VT[c..c+3][r]
        sT[(c + 0) * TILE_M + r] = (__bf16)vq.x;
        sT[(c + 1) * TILE_M + r] = (__bf16)vq.y;
        sT[(c + 2) * TILE_M + r] = (__bf16)vq.z;
        sT[(c + 3) * TILE_M + r] = (__bf16)vq.w;
    }
    __syncthreads();
    // coalesced b128 stores of the transposed tile
    const char* src = (const char*)sT;
    char* dst = (char*)(wsVT + tileBase);
#pragma unroll
    for (int j = 0; j < 4; ++j) {
        const int off = tid * 64 + j * 16;
        *(uint4*)(dst + off) = *(const uint4*)(src + off);
    }
}

// =============================================================================
// Fast path: bf16 K / V^T tiles async-copied into double-buffered LDS.
// =============================================================================
__global__ __launch_bounds__(WAVES * 32)
void attn_fwd_async_kernel(const float* __restrict__ Q,
                           const __bf16* __restrict__ wsK,
                           const __bf16* __restrict__ wsVT,
                           float* __restrict__ Out,
                           float* __restrict__ Attn)
{
    __shared__ __align__(64) __bf16 sK[2][TILE_ELEMS];   // 2 x 8KB
    __shared__ __align__(64) __bf16 sVT[2][TILE_ELEMS];  // 2 x 8KB
    __shared__ __align__(64) float  sP[WAVES][16 * TILE_M];

    const int tid  = threadIdx.x;
    const int wave = tid >> 5;
    const int lane = tid & 31;
    const int ln16 = lane & 15;
    const int half = lane >> 4;

    // uniform batch (from blockIdx only, so SGPR-safe for async saddr)
    const int b     = blockIdx.x >> 5;                   // 32 blocks per batch
    const int qrow0 = ((blockIdx.x & 31) * WAVES + wave) * 16;

    const float scale = 0.125f;

    // ---- Q as WMMA A operands ----
    const float* qrow = Q + ((size_t)b * SEQ + qrow0 + ln16) * DK;
    const int ka = half * 16;
    v16bf qa0, qa1;
#pragma unroll
    for (int j = 0; j < 8; ++j) {
        qa0[j]     = (__bf16)(qrow[ka + j]           * scale);
        qa0[j + 8] = (__bf16)(qrow[ka + 8 + j]       * scale);
        qa1[j]     = (__bf16)(qrow[32 + ka + j]      * scale);
        qa1[j + 8] = (__bf16)(qrow[32 + ka + 8 + j]  * scale);
    }

    // ---- async tile copy: 8KB via 4x global_load_async_to_lds_b128 per thread --
    // INST_OFFSET is added to BOTH the LDS address and the global address, so a
    // single per-thread LDS-addr VGPR + offset VGPR covers all 4 chunks.
    const unsigned voff = (unsigned)(tid * 16);
    auto copyK = [&](int mt, int buf) {
        const __bf16* src = wsK + ((size_t)b * NTILES + mt) * TILE_ELEMS;
        unsigned lds = (unsigned)(size_t)(&sK[buf][0]) + voff;
        asm volatile("global_load_async_to_lds_b128 %0, %1, %2 offset:0"
                     :: "v"(lds), "v"(voff), "s"(src) : "memory");
        asm volatile("global_load_async_to_lds_b128 %0, %1, %2 offset:2048"
                     :: "v"(lds), "v"(voff), "s"(src) : "memory");
        asm volatile("global_load_async_to_lds_b128 %0, %1, %2 offset:4096"
                     :: "v"(lds), "v"(voff), "s"(src) : "memory");
        asm volatile("global_load_async_to_lds_b128 %0, %1, %2 offset:6144"
                     :: "v"(lds), "v"(voff), "s"(src) : "memory");
    };
    auto copyV = [&](int mt, int buf) {
        const __bf16* src = wsVT + ((size_t)b * NTILES + mt) * TILE_ELEMS;
        unsigned lds = (unsigned)(size_t)(&sVT[buf][0]) + voff;
        asm volatile("global_load_async_to_lds_b128 %0, %1, %2 offset:0"
                     :: "v"(lds), "v"(voff), "s"(src) : "memory");
        asm volatile("global_load_async_to_lds_b128 %0, %1, %2 offset:2048"
                     :: "v"(lds), "v"(voff), "s"(src) : "memory");
        asm volatile("global_load_async_to_lds_b128 %0, %1, %2 offset:4096"
                     :: "v"(lds), "v"(voff), "s"(src) : "memory");
        asm volatile("global_load_async_to_lds_b128 %0, %1, %2 offset:6144"
                     :: "v"(lds), "v"(voff), "s"(src) : "memory");
    };

    auto computeS = [&](const __bf16* kbuf, int st) -> v8f {
        const __bf16* kr = kbuf + (st * 16 + ln16) * DK + half * 16;
        const v16bf kb0 = *(const v16bf*)(kr);
        const v16bf kb1 = *(const v16bf*)(kr + 32);
        v8f z = {};
        v8f s = __builtin_amdgcn_wmma_f32_16x16x32_bf16(false, qa0, false, kb0,
                                                        (short)0, z, false, false);
        s     = __builtin_amdgcn_wmma_f32_16x16x32_bf16(false, qa1, false, kb1,
                                                        (short)0, s, false, false);
        return s;
    };

    // ================= pass 1: row max m and row sum-exp l =======================
    v8f m, l;
#pragma unroll
    for (int i = 0; i < 8; ++i) { m[i] = -3.0e38f; l[i] = 0.0f; }

    copyK(0, 0);
    asm volatile("s_wait_asynccnt 0x0" ::: "memory");
    __syncthreads();
    for (int mt = 0; mt < NTILES; ++mt) {
        if (mt + 1 < NTILES) copyK(mt + 1, (mt + 1) & 1);  // overlap with compute
        const __bf16* kbuf = sK[mt & 1];
        v8f s[NSUB];
#pragma unroll
        for (int st = 0; st < NSUB; ++st) s[st] = computeS(kbuf, st);
#pragma unroll
        for (int i = 0; i < 8; ++i) {
            float t = fmaxf(fmaxf(s[0][i], s[1][i]), fmaxf(s[2][i], s[3][i]));
            t = red16_max(t);
            const float mn = fmaxf(m[i], t);
            float r = __expf(s[0][i] - mn) + __expf(s[1][i] - mn)
                    + __expf(s[2][i] - mn) + __expf(s[3][i] - mn);
            r = red16_sum(r);
            l[i] = l[i] * __expf(m[i] - mn) + r;
            m[i] = mn;
        }
        asm volatile("s_wait_asynccnt 0x0" ::: "memory");
        __syncthreads();
    }

    v8f rl;
#pragma unroll
    for (int i = 0; i < 8; ++i) rl[i] = 1.0f / l[i];

    // ================= pass 2: P = exp(S-m)/l, attn store, O += P @ V ============
    v8f o[4];
#pragma unroll
    for (int d = 0; d < 4; ++d) { v8f z = {}; o[d] = z; }

    float* pw = sP[wave];
    copyK(0, 0);
    copyV(0, 0);
    asm volatile("s_wait_asynccnt 0x0" ::: "memory");
    __syncthreads();
    for (int mt = 0; mt < NTILES; ++mt) {
        if (mt + 1 < NTILES) {
            copyK(mt + 1, (mt + 1) & 1);
            copyV(mt + 1, (mt + 1) & 1);
        }
        const __bf16* kbuf = sK[mt & 1];
        const __bf16* vbuf = sVT[mt & 1];

        v8f s[NSUB];
#pragma unroll
        for (int st = 0; st < NSUB; ++st) s[st] = computeS(kbuf, st);
#pragma unroll
        for (int st = 0; st < NSUB; ++st)
#pragma unroll
            for (int i = 0; i < 8; ++i)
                s[st][i] = __expf(s[st][i] - m[i]) * rl[i];

        // C-layout -> row-major f32 P tile (16x64) in per-wave LDS scratch
#pragma unroll
        for (int st = 0; st < NSUB; ++st)
#pragma unroll
            for (int i = 0; i < 8; ++i)
                pw[(i + half * 8) * TILE_M + st * 16 + ln16] = s[st][i];
        asm volatile("s_wait_dscnt 0x0" ::: "memory");     // same-wave LDS RAW

        // coalesced attn stores: 8x b128 per lane
        {
            float* ab = Attn + ((size_t)b * SEQ + qrow0) * SEQ + (size_t)mt * TILE_M;
#pragma unroll
            for (int c = 0; c < 8; ++c) {
                const int w   = lane * 4 + c * 128;
                const int row = w >> 6, col = w & 63;
                *(float4*)(ab + (size_t)row * SEQ + col) = *(const float4*)(pw + w);
            }
        }

        // P as WMMA A operands (bf16 16x32 x2), lane-transposed via LDS
        v16bf pa0, pa1;
        {
            const float* pr = pw + ln16 * TILE_M + half * 16;
#pragma unroll
            for (int j = 0; j < 8; ++j) {
                pa0[j]     = (__bf16)pr[j];
                pa0[j + 8] = (__bf16)pr[8 + j];
                pa1[j]     = (__bf16)pr[32 + j];
                pa1[j + 8] = (__bf16)pr[40 + j];
            }
        }

        // O[16x64] += P[16x64] @ V[64x64]
#pragma unroll
        for (int dblk = 0; dblk < 4; ++dblk) {
            const __bf16* vr = vbuf + (dblk * 16 + ln16) * TILE_M + half * 16;
            const v16bf vb0 = *(const v16bf*)(vr);
            const v16bf vb1 = *(const v16bf*)(vr + 32);
            o[dblk] = __builtin_amdgcn_wmma_f32_16x16x32_bf16(false, pa0, false, vb0,
                                                              (short)0, o[dblk], false, false);
            o[dblk] = __builtin_amdgcn_wmma_f32_16x16x32_bf16(false, pa1, false, vb1,
                                                              (short)0, o[dblk], false, false);
        }
        asm volatile("s_wait_asynccnt 0x0" ::: "memory");
        __syncthreads();
    }

    float* ob = Out + ((size_t)b * SEQ + qrow0) * DK;
#pragma unroll
    for (int dblk = 0; dblk < 4; ++dblk)
#pragma unroll
        for (int i = 0; i < 8; ++i)
            ob[(i + half * 8) * DK + dblk * 16 + ln16] = o[dblk][i];
}

// =============================================================================
// Fallback (self-contained, converts f32->bf16 in-kernel) if ws is too small.
// =============================================================================
__global__ __launch_bounds__(WAVES * 32)
void attn_fwd_kernel(const float* __restrict__ Q, const float* __restrict__ K,
                     const float* __restrict__ V, float* __restrict__ Out,
                     float* __restrict__ Attn)
{
    __shared__ __align__(64) __bf16 sK[TILE_ELEMS];
    __shared__ __align__(64) __bf16 sVT[TILE_ELEMS];
    __shared__ __align__(64) float  sP[WAVES][16 * TILE_M];

    const int tid  = threadIdx.x;
    const int wave = tid >> 5;
    const int lane = tid & 31;
    const int ln16 = lane & 15;
    const int half = lane >> 4;

    const int band  = blockIdx.x * WAVES + wave;
    const int b     = band / (SEQ / 16);
    const int qrow0 = (band % (SEQ / 16)) * 16;
    const float scale = 0.125f;

    const float* qrow = Q + ((size_t)b * SEQ + qrow0 + ln16) * DK;
    const int ka = half * 16;
    v16bf qa0, qa1;
#pragma unroll
    for (int j = 0; j < 8; ++j) {
        qa0[j]     = (__bf16)(qrow[ka + j]           * scale);
        qa0[j + 8] = (__bf16)(qrow[ka + 8 + j]       * scale);
        qa1[j]     = (__bf16)(qrow[32 + ka + j]      * scale);
        qa1[j + 8] = (__bf16)(qrow[32 + ka + 8 + j]  * scale);
    }

    auto stage = [&](int mt, bool withV) {
        const float* kg = K + ((size_t)b * SEQ + mt * TILE_M) * DK;
        const float* vg = V + ((size_t)b * SEQ + mt * TILE_M) * DK;
        if (mt + 1 < NTILES) {
            __builtin_prefetch(kg + TILE_ELEMS + tid * 32, 0, 1);
            if (withV) __builtin_prefetch(vg + TILE_ELEMS + tid * 32, 0, 1);
        }
#pragma unroll
        for (int i = 0; i < 8; ++i) {
            const int e = tid * 4 + i * 512;
            const float4 kq = *(const float4*)(kg + e);
            sK[e + 0] = (__bf16)kq.x; sK[e + 1] = (__bf16)kq.y;
            sK[e + 2] = (__bf16)kq.z; sK[e + 3] = (__bf16)kq.w;
            if (withV) {
                const float4 vq = *(const float4*)(vg + e);
                const int r = e >> 6, c = e & 63;
                sVT[(c + 0) * TILE_M + r] = (__bf16)vq.x;
                sVT[(c + 1) * TILE_M + r] = (__bf16)vq.y;
                sVT[(c + 2) * TILE_M + r] = (__bf16)vq.z;
                sVT[(c + 3) * TILE_M + r] = (__bf16)vq.w;
            }
        }
    };
    auto computeS = [&](int st) -> v8f {
        const __bf16* kr = &sK[(st * 16 + ln16) * DK + half * 16];
        const v16bf kb0 = *(const v16bf*)(kr);
        const v16bf kb1 = *(const v16bf*)(kr + 32);
        v8f z = {};
        v8f s = __builtin_amdgcn_wmma_f32_16x16x32_bf16(false, qa0, false, kb0,
                                                        (short)0, z, false, false);
        s     = __builtin_amdgcn_wmma_f32_16x16x32_bf16(false, qa1, false, kb1,
                                                        (short)0, s, false, false);
        return s;
    };

    v8f m, l;
#pragma unroll
    for (int i = 0; i < 8; ++i) { m[i] = -3.0e38f; l[i] = 0.0f; }

    for (int mt = 0; mt < NTILES; ++mt) {
        stage(mt, false);
        __syncthreads();
        v8f s[NSUB];
#pragma unroll
        for (int st = 0; st < NSUB; ++st) s[st] = computeS(st);
        __syncthreads();
#pragma unroll
        for (int i = 0; i < 8; ++i) {
            float t = fmaxf(fmaxf(s[0][i], s[1][i]), fmaxf(s[2][i], s[3][i]));
            t = red16_max(t);
            const float mn = fmaxf(m[i], t);
            float r = __expf(s[0][i] - mn) + __expf(s[1][i] - mn)
                    + __expf(s[2][i] - mn) + __expf(s[3][i] - mn);
            r = red16_sum(r);
            l[i] = l[i] * __expf(m[i] - mn) + r;
            m[i] = mn;
        }
    }
    v8f rl;
#pragma unroll
    for (int i = 0; i < 8; ++i) rl[i] = 1.0f / l[i];

    v8f o[4];
#pragma unroll
    for (int d = 0; d < 4; ++d) { v8f z = {}; o[d] = z; }
    float* pw = sP[wave];
    for (int mt = 0; mt < NTILES; ++mt) {
        stage(mt, true);
        __syncthreads();
        v8f s[NSUB];
#pragma unroll
        for (int st = 0; st < NSUB; ++st) s[st] = computeS(st);
#pragma unroll
        for (int st = 0; st < NSUB; ++st)
#pragma unroll
            for (int i = 0; i < 8; ++i)
                s[st][i] = __expf(s[st][i] - m[i]) * rl[i];
#pragma unroll
        for (int st = 0; st < NSUB; ++st)
#pragma unroll
            for (int i = 0; i < 8; ++i)
                pw[(i + half * 8) * TILE_M + st * 16 + ln16] = s[st][i];
        asm volatile("s_wait_dscnt 0x0" ::: "memory");
        {
            float* ab = Attn + ((size_t)b * SEQ + qrow0) * SEQ + (size_t)mt * TILE_M;
#pragma unroll
            for (int c = 0; c < 8; ++c) {
                const int w   = lane * 4 + c * 128;
                const int row = w >> 6, col = w & 63;
                *(float4*)(ab + (size_t)row * SEQ + col) = *(const float4*)(pw + w);
            }
        }
        v16bf pa0, pa1;
        {
            const float* pr = pw + ln16 * TILE_M + half * 16;
#pragma unroll
            for (int j = 0; j < 8; ++j) {
                pa0[j]     = (__bf16)pr[j];
                pa0[j + 8] = (__bf16)pr[8 + j];
                pa1[j]     = (__bf16)pr[32 + j];
                pa1[j + 8] = (__bf16)pr[40 + j];
            }
        }
#pragma unroll
        for (int dblk = 0; dblk < 4; ++dblk) {
            const __bf16* vr = &sVT[(dblk * 16 + ln16) * TILE_M + half * 16];
            const v16bf vb0 = *(const v16bf*)(vr);
            const v16bf vb1 = *(const v16bf*)(vr + 32);
            o[dblk] = __builtin_amdgcn_wmma_f32_16x16x32_bf16(false, pa0, false, vb0,
                                                              (short)0, o[dblk], false, false);
            o[dblk] = __builtin_amdgcn_wmma_f32_16x16x32_bf16(false, pa1, false, vb1,
                                                              (short)0, o[dblk], false, false);
        }
        __syncthreads();
    }
    float* ob = Out + ((size_t)b * SEQ + qrow0) * DK;
#pragma unroll
    for (int dblk = 0; dblk < 4; ++dblk)
#pragma unroll
        for (int i = 0; i < 8; ++i)
            ob[(i + half * 8) * DK + dblk * 16 + ln16] = o[dblk][i];
}

extern "C" void kernel_launch(void* const* d_in, const int* in_sizes, int n_in,
                              void* d_out, int out_size, void* d_ws, size_t ws_size,
                              hipStream_t stream) {
    const float* q = (const float*)d_in[0];
    const float* k = (const float*)d_in[1];
    const float* v = (const float*)d_in[2];
    float* out  = (float*)d_out;
    float* attn = out + (size_t)BATCH * SEQ * DK;

    const int blocks = (BATCH * (SEQ / 16)) / WAVES;    // 512 workgroups

    const size_t need = (size_t)2 * BATCH * SEQ * DK * sizeof(__bf16);  // K + V^T bf16
    if (ws_size >= need) {
        __bf16* wsK  = (__bf16*)d_ws;
        __bf16* wsVT = wsK + (size_t)BATCH * SEQ * DK;
        convert_kv_kernel<<<BATCH * NTILES, 128, 0, stream>>>(k, v, wsK, wsVT);
        attn_fwd_async_kernel<<<blocks, WAVES * 32, 0, stream>>>(q, wsK, wsVT, out, attn);
    } else {
        attn_fwd_kernel<<<blocks, WAVES * 32, 0, stream>>>(q, k, v, out, attn);
    }
}
